// MegalodonAttention_68315749810467
// MI455X (gfx1250) — matrix operand order repaired and checked by reading, not compiled, not run on hardware
//
#include <hip/hip_runtime.h>
#include <hip/hip_bf16.h>
#include <math.h>

typedef __attribute__((ext_vector_type(16))) __bf16        v16bf;
typedef __attribute__((ext_vector_type(8)))  float         v8f;
typedef __attribute__((ext_vector_type(8)))  unsigned int  v8u;

#define NHEADS 16
#define DHEAD  128
#define DVAL   128
#define SEQL   4096
#define CHK    1024
#define QTILE  128
#define QSTR   136   // padded bf16 row stride for Q/K tiles
#define VSTR   40    // padded bf16 row stride for transposed V
#define PSTR   40    // padded bf16 row stride for P scratch
#define OUTD   (NHEADS*DVAL)

__device__ __forceinline__ v8f wmma_bf16(v16bf a, v16bf b, v8f c) {
  // D = A(16x32 bf16) * B(32x16 bf16) + C(16x16 f32)
  return __builtin_amdgcn_wmma_f32_16x16x32_bf16(false, a, false, b, (short)0, c, false, false);
}

// A-fragment (16x32, 16-bit): per lane 16 contiguous K values K{2kb..2kb+15}, kb = 0 (low half) / 8 (high half)
__device__ __forceinline__ v16bf load_a_frag(const __bf16* rowPtr, int kbase) {
  const unsigned int* p0 = (const unsigned int*)(rowPtr + kbase * 2);
  const unsigned int* p1 = (const unsigned int*)(rowPtr + kbase * 2 + 8);
  v8u t;
  t[0] = p0[0]; t[1] = p0[1]; t[2] = p0[2]; t[3] = p0[3];
  t[4] = p1[0]; t[5] = p1[1]; t[6] = p1[2]; t[7] = p1[3];
  return __builtin_bit_cast(v16bf, t);
}

// B-fragment (32x16, 16-bit): per lane 16 contiguous K values (lanes<16: K0..15, lanes>=16: K16..31)
__device__ __forceinline__ v16bf load_b_frag(const __bf16* ptr) {
  const unsigned int* p = (const unsigned int*)ptr;
  v8u t;
#pragma unroll
  for (int i = 0; i < 8; ++i) t[i] = p[i];
  return __builtin_bit_cast(v16bf, t);
}

// XOR-butterfly lane exchange via DS_SWIZZLE_B32 (group-of-32 mode, and_mask=0x1f,
// or_mask=0, xor_mask=MASK). One DS op, no index VALU (vs ds_bpermute from __shfl_xor).
template <int MASK>
__device__ __forceinline__ float swz_xor(float x) {
  return __builtin_bit_cast(float,
      __builtin_amdgcn_ds_swizzle(__builtin_bit_cast(int, x), 0x1f | (MASK << 10)));
}
__device__ __forceinline__ float redmax16(float v) {
  v = fmaxf(v, swz_xor<1>(v));
  v = fmaxf(v, swz_xor<2>(v));
  v = fmaxf(v, swz_xor<4>(v));
  v = fmaxf(v, swz_xor<8>(v));
  return v;
}
__device__ __forceinline__ float redsum16(float v) {
  v += swz_xor<1>(v);
  v += swz_xor<2>(v);
  v += swz_xor<4>(v);
  v += swz_xor<8>(v);
  return v;
}

__global__ __launch_bounds__(256, 1)
void megalodon_flash_fwd(const float* __restrict__ q, const float* __restrict__ k,
                         const float* __restrict__ v, const int* __restrict__ start_idx,
                         float* __restrict__ out) {
  __shared__ __align__(16) __bf16 Qs[QTILE * QSTR];   // Q tile, RoPE'd, bf16 row-major
  __shared__ __align__(16) __bf16 Ks[32 * QSTR];      // K tile (32 keys), RoPE'd, row-major
  __shared__ __align__(16) __bf16 Vs[DVAL * VSTR];    // V tile transposed: [dim][key]
  __shared__ __align__(16) __bf16 Ps[8 * 16 * PSTR];  // per-wave P transpose scratch

  const int bid  = blockIdx.x;
  const int qt   = bid & 7;          // q-tile within chunk
  const int h    = (bid >> 3) & 15;  // head
  const int c    = (bid >> 7) & 3;   // chunk
  const int b    = bid >> 9;         // batch
  const int qbase = qt * QTILE;

  const int tid  = threadIdx.x;
  const int lane = tid & 31;
  const int wv   = tid >> 5;         // wave id 0..7 -> owns q rows qbase+wv*16 ..+15
  const int ln   = lane & 15;
  const int half = lane >> 4;
  const int start = start_idx[0];

  const float kRopeC = -0.14391156831212790f;   // -ln(10000)/64
  const float sc     = 0.08838834764831845f;    // 1/sqrt(128)

  // ---------------- load + RoPE Q tile -> LDS (bf16) ----------------
#pragma unroll 4
  for (int task = tid; task < QTILE * 64; task += 256) {
    const int row = task >> 6;
    const int j   = task & 63;
    const int l   = c * CHK + qbase + row;
    const float* qp = q + (((size_t)b * SEQL + l) * NHEADS + h) * DHEAD;
    const float x1 = qp[j], x2 = qp[j + 64];
    const float fr = __expf((float)j * kRopeC);
    const float ang = (float)(l + start) * fr;
    float sn, cs; __sincosf(ang, &sn, &cs);
    Qs[row * QSTR + j]      = (__bf16)(x1 * cs - x2 * sn);
    Qs[row * QSTR + j + 64] = (__bf16)(x2 * cs + x1 * sn);
  }
  __syncthreads();

  // per-wave Q A-fragments (16 rows x 128 dims = 4 fragments of 16x32)
  v16bf aQ[4];
  {
    const __bf16* qrow = &Qs[(wv * 16 + ln) * QSTR];
#pragma unroll
    for (int f = 0; f < 4; ++f) aQ[f] = load_a_frag(qrow + f * 32, half ? 8 : 0);
  }

  v8f O[8];
  float m_r[8], l_r[8];
  {
    v8f z = {};
#pragma unroll
    for (int t = 0; t < 8; ++t) O[t] = z;
#pragma unroll
    for (int r = 0; r < 8; ++r) { m_r[r] = -INFINITY; l_r[r] = 0.0f; }
  }

  const int iters = 4 * (qt + 1);   // keys 0 .. qbase+127, 32 per iteration

  // ---- software pipeline: raw K/V for next tile preloaded into registers ----
  float kx1[8], kx2[8], vbuf[16];
  auto preload = [&](int kv0n) {
#pragma unroll
    for (int i = 0; i < 8; ++i) {          // K: 32 rows x 64 dim-pairs
      const int task = tid + i * 256;
      const int row = task >> 6, j = task & 63;
      const float* kp = k + (((size_t)b * SEQL + (c * CHK + kv0n + row)) * NHEADS + h) * DHEAD;
      kx1[i] = kp[j];
      kx2[i] = kp[j + 64];
    }
#pragma unroll
    for (int i = 0; i < 16; ++i) {         // V: 32 rows x 128 dims
      const int task = tid + i * 256;
      const int row = task >> 7, d = task & 127;
      vbuf[i] = v[(((size_t)b * SEQL + (c * CHK + kv0n + row)) * NHEADS + h) * DVAL + d];
    }
  };
  preload(0);

  for (int it = 0; it < iters; ++it) {
    const int kv0 = it * 32;
    __syncthreads();   // all waves done reading previous Ks/Vs

    // ---- store phase: RoPE + bf16 convert from register buffers -> LDS ----
#pragma unroll
    for (int i = 0; i < 8; ++i) {
      const int task = tid + i * 256;
      const int row = task >> 6, j = task & 63;
      const int l = c * CHK + kv0 + row;
      const float fr = __expf((float)j * kRopeC);
      const float ang = (float)(l + start) * fr;
      float sn, cs; __sincosf(ang, &sn, &cs);
      Ks[row * QSTR + j]      = (__bf16)(kx1[i] * cs - kx2[i] * sn);
      Ks[row * QSTR + j + 64] = (__bf16)(kx2[i] * cs + kx1[i] * sn);
    }
#pragma unroll
    for (int i = 0; i < 16; ++i) {
      const int task = tid + i * 256;
      const int row = task >> 7, d = task & 127;
      Vs[d * VSTR + row] = (__bf16)vbuf[i];
    }
    __syncthreads();

    // issue next tile's global loads now; they stay in flight during the math below
    if (it + 1 < iters) preload((it + 1) * 32);

    if (kv0 > qbase + wv * 16 + 15) continue;   // fully above diagonal for this wave

    // ---- S = Q K^T (16 x 32), two 16-column WMMA tiles, K-depth 128 ----
    v8f S0 = {}, S1 = {};
#pragma unroll
    for (int f = 0; f < 4; ++f) {
      v16bf b0 = load_b_frag(&Ks[ln * QSTR + f * 32 + half * 16]);
      v16bf b1 = load_b_frag(&Ks[(16 + ln) * QSTR + f * 32 + half * 16]);
      S0 = wmma_bf16(aQ[f], b0, S0);
      S1 = wmma_bf16(aQ[f], b1, S1);
    }

    // ---- scale, causal mask, online softmax (C layout: N=ln, M=r+8*half) ----
    const int qr0 = qbase + wv * 16 + half * 8;
    float p0[8], p1[8], scl[8];
#pragma unroll
    for (int r = 0; r < 8; ++r) {
      float s0 = S0[r] * sc, s1 = S1[r] * sc;
      const int qr = qr0 + r;
      if (kv0 + ln > qr)      s0 = -INFINITY;
      if (kv0 + 16 + ln > qr) s1 = -INFINITY;
      const float mx = redmax16(fmaxf(s0, s1));
      const float mn = fmaxf(m_r[r], mx);
      scl[r] = __expf(m_r[r] - mn);
      const float e0 = __expf(s0 - mn);
      const float e1 = __expf(s1 - mn);
      l_r[r] = l_r[r] * scl[r] + redsum16(e0 + e1);
      m_r[r] = mn;
      p0[r] = e0; p1[r] = e1;
    }
#pragma unroll
    for (int t = 0; t < 8; ++t) {
#pragma unroll
      for (int r = 0; r < 8; ++r) O[t][r] *= scl[r];
    }

    // ---- C-layout -> A-layout transpose of P through per-wave LDS scratch ----
    __bf16* pw = &Ps[wv * 16 * PSTR];
#pragma unroll
    for (int r = 0; r < 8; ++r) {
      pw[(r + half * 8) * PSTR + ln]      = (__bf16)p0[r];
      pw[(r + half * 8) * PSTR + 16 + ln] = (__bf16)p1[r];
    }
    __builtin_amdgcn_wave_barrier();   // LDS in-order per wave; block reordering
    const v16bf aP = load_a_frag(&pw[ln * PSTR], half ? 8 : 0);

    // ---- O += P (16x32) @ V (32x128), 8 output column tiles ----
#pragma unroll
    for (int t = 0; t < 8; ++t) {
      v16bf bV = load_b_frag(&Vs[(t * 16 + ln) * VSTR + half * 16]);
      O[t] = wmma_bf16(aP, bV, O[t]);
    }
  }

  // ---------------- epilogue: normalize and store ----------------
#pragma unroll
  for (int r = 0; r < 8; ++r) {
    const float inv = 1.0f / l_r[r];
    const int l = c * CHK + qbase + wv * 16 + half * 8 + r;
    float* op = out + ((size_t)b * SEQL + l) * OUTD + h * DVAL + ln;
#pragma unroll
    for (int t = 0; t < 8; ++t) op[t * 16] = O[t][r] * inv;
  }
}

extern "C" void kernel_launch(void* const* d_in, const int* in_sizes, int n_in,
                              void* d_out, int out_size, void* d_ws, size_t ws_size,
                              hipStream_t stream) {
  (void)in_sizes; (void)n_in; (void)out_size; (void)d_ws; (void)ws_size;
  const float* q = (const float*)d_in[0];
  const float* k = (const float*)d_in[1];
  const float* v = (const float*)d_in[2];
  const int*  si = (const int*)d_in[3];
  float* out = (float*)d_out;
  // grid = B(2) * chunks(4) * heads(16) * qtiles(8) = 1024 workgroups of 256 threads (8 waves)
  megalodon_flash_fwd<<<dim3(1024), dim3(256), 0, stream>>>(q, k, v, si, out);
}
